// MaskedMultiHeadAttention_89429809038088
// MI455X (gfx1250) — matrix operand order repaired and checked by reading, not compiled, not run on hardware
//
#include <hip/hip_runtime.h>
#include <cstdint>

// ---------------------------------------------------------------------------
// MaskedMultiHeadAttention for MI455X (gfx1250, wave32, WMMA bf16 16x16x32)
// bs=2, seq=2048, d_model=1024, n_head=16, dh=64
// ---------------------------------------------------------------------------

#define D_MODEL 1024
#define N_HEAD  16
#define DH      64
#define BS      2
#define SEQ     2048
#define ROWS    (BS * SEQ)      // 4096
#define NQKV    (3 * D_MODEL)   // 3072

typedef __attribute__((ext_vector_type(8)))  __bf16 bf16x8;
typedef __attribute__((ext_vector_type(16))) __bf16 bf16x16;
typedef __attribute__((ext_vector_type(8)))  float  f32x8;

union AFrag { bf16x16 v; bf16x8 h[2]; };

__device__ __forceinline__ unsigned short f2bf(float f) {
  unsigned int u = __float_as_uint(f);
  u += 0x7fffu + ((u >> 16) & 1u);        // round-to-nearest-even
  return (unsigned short)(u >> 16);
}
__device__ __forceinline__ unsigned int pack2(float a, float b) {
  return (unsigned int)f2bf(a) | ((unsigned int)f2bf(b) << 16);
}
__device__ __forceinline__ f32x8 wmma_bf16(bf16x16 a, bf16x16 b, f32x8 c) {
  return __builtin_amdgcn_wmma_f32_16x16x32_bf16(false, a, false, b,
                                                 (short)0, c, false, false);
}

// ---------------------------------------------------------------------------
// 1) f32 -> bf16 elementwise convert (x), 4 elements / thread
// ---------------------------------------------------------------------------
__global__ void cvt_kernel(const float* __restrict__ in,
                           unsigned short* __restrict__ out, int n4) {
  const int i = blockIdx.x * blockDim.x + threadIdx.x;
  if (i >= n4) return;
  const float4 f = ((const float4*)in)[i];
  uint2 o;
  o.x = pack2(f.x, f.y);
  o.y = pack2(f.z, f.w);
  ((uint2*)out)[i] = o;
}

// ---------------------------------------------------------------------------
// 2) transpose + convert: W [K][N] f32 -> WT [N][K] bf16 (LDS 32x32 tiles)
// ---------------------------------------------------------------------------
__global__ void transpose_cvt_kernel(const float* __restrict__ W,
                                     unsigned short* __restrict__ WT,
                                     int K, int N) {
  __shared__ float tile[32][33];
  const int n0 = blockIdx.x * 32;
  const int k0 = blockIdx.y * 32;
  const int tx = threadIdx.x, ty = threadIdx.y;
  for (int i = 0; i < 32; i += 8)
    tile[ty + i][tx] = W[(size_t)(k0 + ty + i) * N + (n0 + tx)];
  __syncthreads();
  for (int i = 0; i < 32; i += 8)
    WT[(size_t)(n0 + ty + i) * K + (k0 + tx)] = f2bf(tile[tx][ty + i]);
}

// ---------------------------------------------------------------------------
// 3) QKV GEMM: Xb[4096][1024]bf16 @ WTq[3072][1024]bf16 + bias
//    block tile = 32(M) x 512(N); 8 waves share one 32-row A stripe that is
//    staged per k-step into LDS with CDNA5 async global->LDS copies
//    (ASYNCcnt), then read back as WMMA A-fragments via ds_load_b128.
//    outputs: Q,K -> [b,h,s,dh], V -> [b,h,dh,s]
// ---------------------------------------------------------------------------
__global__ void __launch_bounds__(256) qkv_gemm_kernel(
    const unsigned short* __restrict__ Xb,
    const unsigned short* __restrict__ WTq,
    const float* __restrict__ bqkv,
    unsigned short* __restrict__ Qh,
    unsigned short* __restrict__ Kh,
    unsigned short* __restrict__ VTh) {
  __shared__ __align__(16) unsigned short atile[32 * 32];   // 2 KB per k-step
  const int tid  = threadIdx.x;
  const int lane = tid & 31;
  const int wave = tid >> 5;
  const int NBLK = NQKV / 512;                // 6
  const int M0   = (blockIdx.x / NBLK) * 32;
  const int N0   = (blockIdx.x % NBLK) * 512 + wave * 64;
  const int lr   = lane & 15;
  const int hi   = lane >> 4;                 // 0|1
  const int kbA  = hi * 8;
  const int kbB  = hi * 16;

  // async-copy mapping: 256 threads x 8 bytes = 2 KB tile
  const int ar = tid >> 3;                    // row 0..31
  const int ac = (tid & 7) * 4;               // bf16 col 0,4,...,28
  const unsigned lds_dst =
      (unsigned)(uintptr_t)(const void*)&atile[ar * 32 + ac];

  const f32x8 zero = {};
  f32x8 acc[2][4];
  for (int m = 0; m < 2; ++m)
    for (int n = 0; n < 4; ++n) acc[m][n] = zero;

  for (int k0 = 0; k0 < D_MODEL; k0 += 32) {
    // ---- stage the shared A stripe (32 rows x 32 k) into LDS, async ----
    {
      const unsigned short* gsrc = Xb + (size_t)(M0 + ar) * D_MODEL + k0 + ac;
      asm volatile("global_load_async_to_lds_b64 %0, %1, off"
                   :: "v"(lds_dst), "v"(gsrc) : "memory");
      asm volatile("s_wait_asynccnt 0" ::: "memory");
    }
    __syncthreads();

    AFrag a[2];
    for (int m = 0; m < 2; ++m) {
      const unsigned short* p = &atile[(m * 16 + lr) * 32 + kbA];
      a[m].h[0] = *(const bf16x8*)p;
      a[m].h[1] = *(const bf16x8*)(p + 16);
    }
    for (int n = 0; n < 4; ++n) {
      const unsigned short* p = WTq + (size_t)(N0 + n * 16 + lr) * D_MODEL + k0 + kbB;
      AFrag bu;
      bu.h[0] = *(const bf16x8*)(p - kbB + kbA);
      bu.h[1] = *(const bf16x8*)(p - kbB + 16 + kbA);
      const bf16x16 bfr = bu.v;
      for (int m = 0; m < 2; ++m)
        acc[m][n] = wmma_bf16(a[m].v, bfr, acc[m][n]);
    }
    __syncthreads();                          // protect atile before next fill
  }

  for (int n = 0; n < 4; ++n) {
    const int c    = N0 + n * 16 + lr;
    const float bias = bqkv[c];
    const int seg  = c >> 10;                 // 0=q 1=k 2=v
    const int cc   = c & 1023;
    const int hd   = cc >> 6;
    const int d    = cc & 63;
    for (int m = 0; m < 2; ++m) {
      for (int v = 0; v < 8; ++v) {
        const int row = M0 + m * 16 + v + hi * 8;
        const int bb  = row >> 11;            // /2048
        const int s   = row & 2047;
        const unsigned short val = f2bf(acc[m][n][v] + bias);
        const size_t bh = (size_t)(bb * N_HEAD + hd);
        if (seg == 0)      Qh [(bh * SEQ + s) * DH + d] = val;
        else if (seg == 1) Kh [(bh * SEQ + s) * DH + d] = val;
        else               VTh[(bh * DH + d) * SEQ + s] = val;
      }
    }
  }
}

// ---------------------------------------------------------------------------
// 4) Flash attention, computed transposed: S^T = K*Q^T, Z^T = V^T*P^T
//    one wave per (b,h,16-query tile); queries live in the lane dimension.
//    Next k-tile K/V^T lines are prefetched (global_prefetch_b8) since the
//    dynamic loop + LDS round-trip prevents cross-iteration pipelining.
// ---------------------------------------------------------------------------
__global__ void __launch_bounds__(256) attn_kernel(
    const unsigned short* __restrict__ Qh,
    const unsigned short* __restrict__ Kh,
    const unsigned short* __restrict__ VTh,
    unsigned short* __restrict__ Zb) {
  __shared__ __align__(32) unsigned short plds[8 * 16 * 32];   // 8 KB
  const int lane = threadIdx.x & 31;
  const int wave = threadIdx.x >> 5;
  const int wg   = blockIdx.x * 8 + wave;
  const int QT   = SEQ / 16;                  // 128
  const int qt   = wg % QT;
  const int bh   = wg / QT;                   // 0..31
  const int bb   = bh >> 4, hd = bh & 15;
  const int Q0   = qt * 16;
  const int lr   = lane & 15;
  const int hi   = lane >> 4;
  const int kbA  = hi * 8;
  const int kbB  = hi * 16;
  const int qg   = Q0 + lr;                   // this lane's query index

  const unsigned short* Qp  = Qh  + (size_t)bh * SEQ * DH;
  const unsigned short* Kp  = Kh  + (size_t)bh * SEQ * DH;
  const unsigned short* VTp = VTh + (size_t)bh * DH * SEQ;
  unsigned short* lds = plds + wave * (16 * 32);

  // Q^T B-fragments, one per 32-wide dh chunk (loaded once)
  AFrag bq[2];
  for (int c = 0; c < 2; ++c) {
    bq[c].h[0] = *(const bf16x8*)(Qp + (size_t)qg * DH + c * 32 + kbA);
    bq[c].h[1] = *(const bf16x8*)(Qp + (size_t)qg * DH + c * 32 + 16 + kbA);
  }

  const f32x8 zero = {};
  f32x8 z[4];
  for (int t = 0; t < 4; ++t) z[t] = zero;
  float m_run = -1e30f, l_run = 0.0f;

  const int kmax = Q0 + 16;
  for (int k0 = 0; k0 < kmax; k0 += 32) {
    // ---- prefetch next k-tile (hide L2 latency behind softmax math) ----
    if (k0 + 32 < kmax) {
      __builtin_prefetch(Kp  + (size_t)(k0 + 32 + lane) * DH, 0, 0);
      __builtin_prefetch(VTp + (size_t)lane        * SEQ + k0 + 32, 0, 0);
      __builtin_prefetch(VTp + (size_t)(lane + 32) * SEQ + k0 + 32, 0, 0);
    }
    // ---- S^T = K * Q^T   (32 keys x 16 queries, f32 accum) ----
    f32x8 s[2];
    s[0] = zero; s[1] = zero;
    for (int j = 0; j < 2; ++j) {
      const unsigned short* kp = Kp + (size_t)(k0 + j * 16 + lr) * DH;
      for (int c = 0; c < 2; ++c) {
        AFrag a;
        a.h[0] = *(const bf16x8*)(kp + c * 32 + kbA);
        a.h[1] = *(const bf16x8*)(kp + c * 32 + 16 + kbA);
        s[j] = wmma_bf16(a.v, bq[c].v, s[j]);
      }
    }
    // ---- scale + causal mask ----
    float sv[2][8];
    float mloc = -1e30f;
    for (int j = 0; j < 2; ++j)
      for (int v = 0; v < 8; ++v) {
        const int key = k0 + j * 16 + v + hi * 8;
        float tv = s[j][v] * 0.125f;           // 1/sqrt(64)
        tv = (key <= qg) ? tv : -1e30f;
        sv[j][v] = tv;
        mloc = fmaxf(mloc, tv);
      }
    // ---- online softmax (row = query = lane column) ----
    const float mrow  = fmaxf(mloc, __shfl_xor(mloc, 16, 32));
    const float m_new = fmaxf(m_run, mrow);
    const float resc  = __expf(m_run - m_new);
    float lloc = 0.0f;
    for (int j = 0; j < 2; ++j)
      for (int v = 0; v < 8; v += 2) {
        const float p0 = __expf(sv[j][v]     - m_new);
        const float p1 = __expf(sv[j][v + 1] - m_new);
        lloc += p0 + p1;
        const int kl = j * 16 + v + hi * 8;    // even -> dword aligned
        *(unsigned int*)(lds + lr * 32 + kl) = pack2(p0, p1);
      }
    l_run = l_run * resc + (lloc + __shfl_xor(lloc, 16, 32));
    m_run = m_new;
    for (int t = 0; t < 4; ++t)
      for (int v = 0; v < 8; ++v) z[t][v] *= resc;
    __asm__ volatile("" ::: "memory");          // keep ds store->load order
    // ---- P^T re-read in B-fragment layout (per-wave LDS, in-order) ----
    AFrag pbu;
    pbu.h[0] = *(const bf16x8*)(lds + lr * 32 + kbA);
    pbu.h[1] = *(const bf16x8*)(lds + lr * 32 + 16 + kbA);
    const bf16x16 pb = pbu.v;
    // ---- Z^T += V^T * P^T ----
    for (int t = 0; t < 4; ++t) {
      const unsigned short* vp = VTp + (size_t)(t * 16 + lr) * SEQ + k0;
      AFrag a;
      a.h[0] = *(const bf16x8*)(vp + kbA);
      a.h[1] = *(const bf16x8*)(vp + 16 + kbA);
      z[t] = wmma_bf16(a.v, pb, z[t]);
    }
  }

  // ---- normalize and store Z as [b, s, h*dh] bf16 ----
  const float inv_l = 1.0f / l_run;
  unsigned short* zp = Zb + (size_t)(bb * SEQ + qg) * D_MODEL + hd * DH;
  for (int t = 0; t < 4; ++t)
    for (int v = 0; v < 8; v += 2) {
      const int d = t * 16 + v + hi * 8;        // even -> dword aligned
      *(unsigned int*)(zp + d) = pack2(z[t][v] * inv_l, z[t][v + 1] * inv_l);
    }
}

// ---------------------------------------------------------------------------
// 5) Projection GEMM: Zb[4096][1024]bf16 @ WTp[1024][1024]bf16 + bias -> f32
// ---------------------------------------------------------------------------
__global__ void __launch_bounds__(256) proj_gemm_kernel(
    const unsigned short* __restrict__ Zb,
    const unsigned short* __restrict__ WTp,
    const float* __restrict__ bproj,
    float* __restrict__ out) {
  const int lane = threadIdx.x & 31;
  const int wave = threadIdx.x >> 5;
  const int wg   = blockIdx.x * 8 + wave;
  const int NT   = D_MODEL / 64;              // 16
  const int M0   = (wg / NT) * 32;
  const int N0   = (wg % NT) * 64;
  const int lr   = lane & 15;
  const int hi   = lane >> 4;
  const int kbA  = hi * 8;
  const int kbB  = hi * 16;

  const f32x8 zero = {};
  f32x8 acc[2][4];
  for (int m = 0; m < 2; ++m)
    for (int n = 0; n < 4; ++n) acc[m][n] = zero;

  for (int k0 = 0; k0 < D_MODEL; k0 += 32) {
    AFrag a[2];
    for (int m = 0; m < 2; ++m) {
      const unsigned short* p = Zb + (size_t)(M0 + m * 16 + lr) * D_MODEL + k0 + kbA;
      a[m].h[0] = *(const bf16x8*)p;
      a[m].h[1] = *(const bf16x8*)(p + 16);
    }
    for (int n = 0; n < 4; ++n) {
      const unsigned short* p = WTp + (size_t)(N0 + n * 16 + lr) * D_MODEL + k0 + kbB;
      AFrag bu;
      bu.h[0] = *(const bf16x8*)(p - kbB + kbA);
      bu.h[1] = *(const bf16x8*)(p - kbB + 16 + kbA);
      const bf16x16 bfr = bu.v;
      for (int m = 0; m < 2; ++m)
        acc[m][n] = wmma_bf16(a[m].v, bfr, acc[m][n]);
    }
  }

  for (int n = 0; n < 4; ++n) {
    const int c = N0 + n * 16 + lr;
    const float bias = bproj[c];
    for (int m = 0; m < 2; ++m)
      for (int v = 0; v < 8; ++v) {
        const int row = M0 + m * 16 + v + hi * 8;
        out[(size_t)row * D_MODEL + c] = acc[m][n][v] + bias;
      }
  }
}

// ---------------------------------------------------------------------------
// Launcher
// ---------------------------------------------------------------------------
extern "C" void kernel_launch(void* const* d_in, const int* in_sizes, int n_in,
                              void* d_out, int out_size, void* d_ws, size_t ws_size,
                              hipStream_t stream) {
  const float* x     = (const float*)d_in[0];  // [2,2048,1024]
  const float* Wqkv  = (const float*)d_in[1];  // [1024,3072]
  const float* bqkv  = (const float*)d_in[2];  // [3072]
  const float* Wproj = (const float*)d_in[3];  // [1024,1024]
  const float* bproj = (const float*)d_in[4];  // [1024]
  float* out = (float*)d_out;                  // [2,2048,1024] f32

  char* ws = (char*)d_ws;
  size_t off = 0;
  auto wsAlloc = [&](size_t bytes) -> void* {
    void* p = ws + off;
    off += (bytes + 255) & ~(size_t)255;
    return p;
  };
  unsigned short* Xb  = (unsigned short*)wsAlloc((size_t)ROWS * D_MODEL * 2);     // 8 MB
  unsigned short* WTq = (unsigned short*)wsAlloc((size_t)NQKV * D_MODEL * 2);     // 6 MB
  unsigned short* WTp = (unsigned short*)wsAlloc((size_t)D_MODEL * D_MODEL * 2);  // 2 MB
  unsigned short* Qh  = (unsigned short*)wsAlloc((size_t)ROWS * D_MODEL * 2);     // 8 MB
  unsigned short* Kh  = (unsigned short*)wsAlloc((size_t)ROWS * D_MODEL * 2);     // 8 MB
  unsigned short* VTh = (unsigned short*)wsAlloc((size_t)ROWS * D_MODEL * 2);     // 8 MB
  unsigned short* Zb  = (unsigned short*)wsAlloc((size_t)ROWS * D_MODEL * 2);     // 8 MB

  // 1) x -> bf16
  {
    const int n4 = ROWS * D_MODEL / 4;
    cvt_kernel<<<(n4 + 255) / 256, 256, 0, stream>>>(x, Xb, n4);
  }
  // 2) weight transpose+convert
  transpose_cvt_kernel<<<dim3(NQKV / 32, D_MODEL / 32), dim3(32, 8), 0, stream>>>(
      Wqkv, WTq, D_MODEL, NQKV);
  transpose_cvt_kernel<<<dim3(D_MODEL / 32, D_MODEL / 32), dim3(32, 8), 0, stream>>>(
      Wproj, WTp, D_MODEL, D_MODEL);
  // 3) QKV GEMM: (4096/32) * (3072/512) blocks, 8 waves each
  qkv_gemm_kernel<<<(ROWS / 32) * (NQKV / 512), 256, 0, stream>>>(
      Xb, WTq, bqkv, Qh, Kh, VTh);
  // 4) attention: 2*16*128 wave tiles / 8
  attn_kernel<<<(BS * N_HEAD * (SEQ / 16)) / 8, 256, 0, stream>>>(Qh, Kh, VTh, Zb);
  // 5) projection GEMM: (4096/32)*(1024/64)/8
  proj_gemm_kernel<<<(ROWS / 32) * (D_MODEL / 64) / 8, 256, 0, stream>>>(
      Zb, WTp, bproj, out);
}
